// ConditionalSmilesRnn_90563680403510
// MI455X (gfx1250) — compile-verified, matrix operand order checked by reading, not run on hardware
//
#include <hip/hip_runtime.h>
#include <stdint.h>

// ---------------------------------------------------------------------------
// Problem dims (fixed by the reference)
// ---------------------------------------------------------------------------
#define Bsz   256
#define Tlen  100
#define STEPS 99
#define Hd    1024
#define Vd    47
#define G4H   4096          // 4*H
#define BH    (Bsz * Hd)    // 262144

typedef __attribute__((ext_vector_type(16))) __bf16 v16bf;
typedef __attribute__((ext_vector_type(8)))  __bf16 v8bf;
typedef __attribute__((ext_vector_type(8)))  float  v8f;

// Load a CDNA5 bf16 WMMA fragment (A or B^T layout): lane holds 8 contiguous
// K values at +0 and 8 more at +16 (caller applies the lane-dependent kLo).
__device__ __forceinline__ v16bf load_frag(const __bf16* p) {
    v8bf lo = *(const v8bf*)(p);
    v8bf hi = *(const v8bf*)(p + 16);
    v16bf r;
#pragma unroll
    for (int i = 0; i < 8; ++i) { r[i] = lo[i]; r[i + 8] = hi[i]; }
    return r;
}

// Async Global -> LDS DMA (gfx1250), tracked by ASYNCcnt.
// Low 32 bits of a generic shared pointer are the LDS byte offset.
__device__ __forceinline__ void async_copy_b128(const __bf16* gsrc, void* ldst) {
    unsigned l = (unsigned)(uintptr_t)ldst;
    asm volatile("global_load_async_to_lds_b128 %0, %1, off"
                 :: "v"(l), "v"(gsrc)
                 : "memory");
}
__device__ __forceinline__ void wait_async0() {
    asm volatile("s_wait_asynccnt 0x0" ::: "memory");
}

__device__ __forceinline__ float fast_exp(float x) {
    return __builtin_amdgcn_exp2f(x * 1.4426950408889634f);
}
__device__ __forceinline__ float sigmoid_f(float x) {
    return __builtin_amdgcn_rcpf(1.0f + fast_exp(-x));
}
__device__ __forceinline__ float tanh_f(float x) {
    float e = fast_exp(2.0f * x);
    return (e - 1.0f) * __builtin_amdgcn_rcpf(e + 1.0f);
}

// ---------------------------------------------------------------------------
// One LSTM layer, one timestep.
//   gates = xA @ Wx^T + hA @ Wh^T + bias ; cell update ; h out (bf16)
// Grid: (H/32, B/128); block: 256 threads = 8 waves.
// Each block owns gate columns [nBase, nBase+32) of all 4 gates.
// Wave w owns rows [blockIdx.y*128 + w*16, +16).
// K loop: 64 iterations (32 for x @ Wx^T, 32 for h @ Wh^T), with async
// Global->LDS staging of the next 8 KB weight tile double-buffered against
// the 8 WMMAs of the current tile.
// ---------------------------------------------------------------------------
__global__ __launch_bounds__(256)
void lstm_step_kernel(const __bf16* __restrict__ Wx,     // [4096][1024] bf16
                      const __bf16* __restrict__ Wh,     // [4096][1024] bf16
                      const __bf16* __restrict__ xA,     // [256][1024] or E_bf
                      const int*    __restrict__ tokens, // x (int32) or null
                      int step,
                      const __bf16* __restrict__ hA,     // prev h, [256][1024]
                      const float*  __restrict__ biasB,  // [256][4096] or [4096]
                      int biasPerRow,
                      float*  __restrict__ cState,       // [256][1024] f32
                      __bf16* __restrict__ hOut,         // [256][1024] bf16
                      __bf16* __restrict__ outsOut)      // null or [256][1024]
{
    __shared__ __align__(16) __bf16 Bs[2][4][32][32];  // double buffer, 2 x 8 KB

    const int tid   = threadIdx.x;
    const int wave  = tid >> 5;
    const int lane  = tid & 31;
    const int rowBase = blockIdx.y * 128 + wave * 16;
    const int nBase   = blockIdx.x * 32;
    const int mLane   = lane & 15;
    const int brow    = rowBase + mLane;
    const int kLo     = (lane & 16) ? 8 : 0;   // CDNA5 bf16 A/B lane->K split

    // A-row base pointers (layer 0 gathers embedding rows by token id)
    const __bf16* arowX;
    if (tokens) {
        int tok = (step == 0) ? 1 : tokens[brow * Tlen + step];
        arowX = xA + (size_t)tok * Hd;
    } else {
        arowX = xA + (size_t)brow * Hd;
    }
    const __bf16* arowH = hA + (size_t)brow * Hd;

    // Each thread DMAs two 16-byte chunks of the 8 KB weight tile.
    const int c0 = tid, c1 = tid + 256;
    const int g0 = c0 >> 7, n0 = (c0 & 127) >> 2, kc0 = c0 & 3;
    const int g1 = c1 >> 7, n1 = (c1 & 127) >> 2, kc1 = c1 & 3;
    const size_t wro0 = (size_t)(g0 * Hd + nBase + n0) * Hd + kc0 * 8;
    const size_t wro1 = (size_t)(g1 * Hd + nBase + n1) * Hd + kc1 * 8;

    auto stage = [&](int ki, int buf) {
        const __bf16* W = (ki < 32) ? Wx : Wh;
        const int k0 = (ki & 31) * 32;
        async_copy_b128(W + wro0 + k0, &Bs[buf][g0][n0][kc0 * 8]);
        async_copy_b128(W + wro1 + k0, &Bs[buf][g1][n1][kc1 * 8]);
    };

    v8f acc[4][2] = {};

    stage(0, 0);
    wait_async0();
    __syncthreads();

#pragma unroll 1
    for (int ki = 0; ki < 64; ++ki) {
        const int cur = ki & 1;
        if (ki < 63) stage(ki + 1, cur ^ 1);   // DMA next tile into other buffer

        const __bf16* arow = (ki < 32) ? arowX : arowH;
        const int k0 = (ki & 31) * 32;
        v16bf a = load_frag(arow + k0 + kLo);
        __builtin_prefetch(arow + k0 + 64, 0, 1);  // global_prefetch_b8

        v16bf bfr[8];
#pragma unroll
        for (int q = 0; q < 8; ++q)
            bfr[q] = load_frag(&Bs[cur][q >> 1][(q & 1) * 16 + mLane][kLo]);
#pragma unroll
        for (int q = 0; q < 8; ++q)
            acc[q >> 1][q & 1] = __builtin_amdgcn_wmma_f32_16x16x32_bf16(
                false, a, false, bfr[q], (short)0, acc[q >> 1][q & 1], false, false);

        wait_async0();     // our DMA into the other buffer has landed
        __syncthreads();   // everyone done reading cur + staging next
    }

    // Cell update: accumulator layout lane->N, vgpr->M (M += 8 for lanes>=16).
    const int colL = nBase + mLane;
    const int rTop = rowBase + ((lane >> 4) & 1) * 8;
#pragma unroll
    for (int jt = 0; jt < 2; ++jt) {
        const int col = colL + jt * 16;
#pragma unroll
        for (int r = 0; r < 8; ++r) {
            const int row  = rTop + r;
            const size_t ci = (size_t)row * Hd + col;
            float bi, bff, bg, bo;
            if (biasPerRow) {
                const float* pb = biasB + (size_t)row * G4H;
                bi = pb[col]; bff = pb[Hd + col]; bg = pb[2 * Hd + col]; bo = pb[3 * Hd + col];
            } else {
                bi = biasB[col]; bff = biasB[Hd + col]; bg = biasB[2 * Hd + col]; bo = biasB[3 * Hd + col];
            }
            float iv = sigmoid_f(acc[0][jt][r] + bi);
            float fv = sigmoid_f(acc[1][jt][r] + bff);
            float gv = tanh_f(acc[2][jt][r] + bg);
            float ov = sigmoid_f(acc[3][jt][r] + bo);
            float cn = fv * cState[ci] + iv * gv;
            cState[ci] = cn;
            float hv = ov * tanh_f(cn);
            __bf16 hb = (__bf16)hv;
            hOut[ci] = hb;
            if (outsOut) outsOut[ci] = hb;
        }
    }
}

// ---------------------------------------------------------------------------
// Decoder: logits[b,s,:] = outs[s,b,:] @ W_dec^T + b_dec   (V padded 47->48)
// Grid: 198 blocks x 256 threads; 128 rows/block of the 25344-row A matrix.
// ---------------------------------------------------------------------------
__global__ __launch_bounds__(256)
void decoder_kernel(const __bf16* __restrict__ outs,   // [99][256][1024] bf16
                    const __bf16* __restrict__ Wdec,   // [48][1024] bf16 (row 47 = 0)
                    const float*  __restrict__ bdec,   // [47]
                    float* __restrict__ logits)        // [256][99][47]
{
    const int tid   = threadIdx.x;
    const int wave  = tid >> 5;
    const int lane  = tid & 31;
    const int rowBase = blockIdx.x * 128 + wave * 16;
    const int mLane   = lane & 15;
    const int kLo     = (lane & 16) ? 8 : 0;
    const __bf16* arow = outs + (size_t)(rowBase + mLane) * Hd;

    v8f acc[3] = {};
#pragma unroll 1
    for (int kt = 0; kt < 32; ++kt) {
        const int k0 = kt * 32;
        v16bf a = load_frag(arow + k0 + kLo);
        v16bf bfr[3];
#pragma unroll
        for (int jt = 0; jt < 3; ++jt)
            bfr[jt] = load_frag(Wdec + (size_t)(jt * 16 + mLane) * Hd + k0 + kLo);
#pragma unroll
        for (int jt = 0; jt < 3; ++jt)
            acc[jt] = __builtin_amdgcn_wmma_f32_16x16x32_bf16(
                false, a, false, bfr[jt], (short)0, acc[jt], false, false);
    }

    const int rTop = rowBase + ((lane >> 4) & 1) * 8;
#pragma unroll
    for (int jt = 0; jt < 3; ++jt) {
        const int col = jt * 16 + mLane;
        if (col < Vd) {
            const float bv = bdec[col];
#pragma unroll
            for (int r = 0; r < 8; ++r) {
                const int row = rTop + r;          // flat = s*256 + b
                const int s  = row >> 8;
                const int bb = row & 255;
                logits[((size_t)bb * STEPS + s) * Vd + col] = acc[jt][r] + bv;
            }
        }
    }
}

// ---------------------------------------------------------------------------
// Prologue / epilogue helpers
// ---------------------------------------------------------------------------
__global__ void f32_to_bf16_pad(const float* __restrict__ src, __bf16* __restrict__ dst,
                                int srcRows, int srcStride, int cols, long long total)
{
    long long idx = (long long)blockIdx.x * 256 + threadIdx.x;
    if (idx >= total) return;
    int r = (int)(idx / cols);
    int k = (int)(idx % cols);
    float v = (r < srcRows) ? src[(size_t)r * srcStride + k] : 0.0f;
    dst[idx] = (__bf16)v;
}

__global__ void f32_copy(const float* __restrict__ src, float* __restrict__ dst, long long n)
{
    long long i = (long long)blockIdx.x * 256 + threadIdx.x;
    if (i < n) dst[i] = src[i];
}

__global__ void bf16_to_f32(const __bf16* __restrict__ src, float* __restrict__ dst, long long n)
{
    long long i = (long long)blockIdx.x * 256 + threadIdx.x;
    if (i < n) dst[i] = (float)src[i];
}

// pbias[b][j] = b_ih0[j] + b_hh0[j] + sum_p props[b][p] * W_ih0[j][1024+p]
__global__ void pbias_kernel(const float* __restrict__ props,
                             const float* __restrict__ W_ih0,   // [4096][1027]
                             const float* __restrict__ b_ih0,
                             const float* __restrict__ b_hh0,
                             float* __restrict__ pbias)
{
    int idx = blockIdx.x * 256 + threadIdx.x;
    if (idx >= Bsz * G4H) return;
    int b = idx >> 12, j = idx & 4095;
    const float* wrow = W_ih0 + (size_t)j * 1027 + 1024;
    float v = b_ih0[j] + b_hh0[j]
            + props[b * 3 + 0] * wrow[0]
            + props[b * 3 + 1] * wrow[1]
            + props[b * 3 + 2] * wrow[2];
    pbias[idx] = v;
}

__global__ void addvec_kernel(const float* a, const float* b, float* o, int n)
{
    int i = blockIdx.x * 256 + threadIdx.x;
    if (i < n) o[i] = a[i] + b[i];
}

// ---------------------------------------------------------------------------
// Host launcher
// ---------------------------------------------------------------------------
extern "C" void kernel_launch(void* const* d_in, const int* in_sizes, int n_in,
                              void* d_out, int out_size, void* d_ws, size_t ws_size,
                              hipStream_t stream)
{
    (void)in_sizes; (void)n_in; (void)out_size; (void)ws_size;

    const int*   x      = (const int*)  d_in[0];
    const float* props  = (const float*)d_in[1];
    const float* h0     = (const float*)d_in[2];
    const float* c0     = (const float*)d_in[3];
    const float* E      = (const float*)d_in[4];
    const float* W_ih0  = (const float*)d_in[5];
    const float* W_hh0  = (const float*)d_in[6];
    const float* b_ih0  = (const float*)d_in[7];
    const float* b_hh0  = (const float*)d_in[8];
    const float* W_ih_r = (const float*)d_in[9];
    const float* W_hh_r = (const float*)d_in[10];
    const float* b_ih_r = (const float*)d_in[11];
    const float* b_hh_r = (const float*)d_in[12];
    const float* W_dec  = (const float*)d_in[13];
    const float* b_dec  = (const float*)d_in[14];

    char* ws = (char*)d_ws;
    size_t off = 0;
    auto alloc = [&](size_t bytes) -> void* {
        void* p = ws + off;
        off = (off + bytes + 255) & ~(size_t)255;
        return p;
    };

    __bf16* W0x   = (__bf16*)alloc((size_t)G4H * Hd * 2);       // W_ih0 emb part
    __bf16* W0h   = (__bf16*)alloc((size_t)G4H * Hd * 2);       // W_hh0
    __bf16* Wxr   = (__bf16*)alloc((size_t)2 * G4H * Hd * 2);   // W_ih_r
    __bf16* Whr   = (__bf16*)alloc((size_t)2 * G4H * Hd * 2);   // W_hh_r
    __bf16* Ebf   = (__bf16*)alloc((size_t)Vd * Hd * 2);
    __bf16* Wdb   = (__bf16*)alloc((size_t)48 * Hd * 2);        // padded decoder
    float*  pbias = (float*) alloc((size_t)Bsz * G4H * 4);
    float*  biasr = (float*) alloc((size_t)2 * G4H * 4);
    __bf16* hbuf0 = (__bf16*)alloc((size_t)3 * BH * 2);
    __bf16* hbuf1 = (__bf16*)alloc((size_t)3 * BH * 2);
    float*  cws   = (float*) alloc((size_t)3 * BH * 4);
    __bf16* outs  = (__bf16*)alloc((size_t)STEPS * BH * 2);

    auto cvt = [&](const float* s, __bf16* d, int srcRows, int srcStride,
                   int cols, long long total) {
        unsigned blocks = (unsigned)((total + 255) / 256);
        f32_to_bf16_pad<<<dim3(blocks), dim3(256), 0, stream>>>(s, d, srcRows, srcStride, cols, total);
    };

    // Weight / state conversion to bf16 (+ decoder padding to 48 rows)
    cvt(E,      Ebf, Vd,   Hd,   Hd, (long long)Vd * Hd);
    cvt(W_ih0,  W0x, G4H,  1027, Hd, (long long)G4H * Hd);   // drop props cols
    cvt(W_hh0,  W0h, G4H,  Hd,   Hd, (long long)G4H * Hd);
    cvt(W_ih_r, Wxr, 2*G4H, Hd,  Hd, (long long)2 * G4H * Hd);
    cvt(W_hh_r, Whr, 2*G4H, Hd,  Hd, (long long)2 * G4H * Hd);
    cvt(W_dec,  Wdb, Vd,   Hd,   Hd, (long long)48 * Hd);    // row 47 zero-padded
    cvt(h0,     hbuf0, 3 * Bsz, Hd, Hd, (long long)3 * BH);

    f32_copy<<<dim3((3 * BH + 255) / 256), dim3(256), 0, stream>>>(c0, cws, (long long)3 * BH);
    pbias_kernel<<<dim3((Bsz * G4H) / 256), dim3(256), 0, stream>>>(props, W_ih0, b_ih0, b_hh0, pbias);
    addvec_kernel<<<dim3((2 * G4H + 255) / 256), dim3(256), 0, stream>>>(b_ih_r, b_hh_r, biasr, 2 * G4H);

    // Recurrent scan: 99 steps x 3 layers (ordering via stream)
    const dim3 grid(Hd / 32, Bsz / 128), blk(256);
    for (int s = 0; s < STEPS; ++s) {
        __bf16* hr = (s & 1) ? hbuf1 : hbuf0;   // read  (prev step)
        __bf16* hw = (s & 1) ? hbuf0 : hbuf1;   // write (this step)

        // layer 0: x-input = embedding gather, bias = per-row pbias
        lstm_step_kernel<<<grid, blk, 0, stream>>>(
            W0x, W0h, Ebf, x, s, hr, pbias, 1, cws, hw, (__bf16*)nullptr);
        // layer 1
        lstm_step_kernel<<<grid, blk, 0, stream>>>(
            Wxr, Whr, hw, (const int*)nullptr, s,
            hr + BH, biasr, 0, cws + BH, hw + BH, (__bf16*)nullptr);
        // layer 2 (also emits the decoder input slice)
        lstm_step_kernel<<<grid, blk, 0, stream>>>(
            Wxr + (size_t)G4H * Hd, Whr + (size_t)G4H * Hd, hw + BH, (const int*)nullptr, s,
            hr + 2 * BH, biasr + G4H, 0, cws + 2 * BH, hw + 2 * BH,
            outs + (size_t)s * BH);
    }

    // Decoder GEMM into d_out (logits region)
    float* out_f = (float*)d_out;
    decoder_kernel<<<dim3((STEPS * Bsz) / 128), blk, 0, stream>>>(outs, Wdb, b_dec, out_f);

    // hT, cT (tuple outputs, concatenated after logits)
    const size_t logitsN = (size_t)Bsz * STEPS * Vd;
    __bf16* hfinal = hbuf1;   // STEPS is odd -> last write landed in hbuf1
    bf16_to_f32<<<dim3((3 * BH + 255) / 256), dim3(256), 0, stream>>>(hfinal, out_f + logitsN, (long long)3 * BH);
    f32_copy<<<dim3((3 * BH + 255) / 256), dim3(256), 0, stream>>>(cws, out_f + logitsN + (size_t)3 * BH, (long long)3 * BH);
}
